// encoderLayer_44006234915475
// MI455X (gfx1250) — matrix operand order repaired and checked by reading, not compiled, not run on hardware
//
#include <hip/hip_runtime.h>
#include <hip/hip_bf16.h>
#include <stdint.h>

// CDNA5 / gfx1250 wave32 WMMA implementation of the encoder layer.
// GEMMs: v_wmma_f32_16x16x32_bf16 (fp32->bf16 inputs, f32 accum).
// Global->LDS staging via Tensor Data Mover (tensor_load_to_lds) when the
// builtin exists; VGPR-staging fallback otherwise.

typedef __bf16 bf16_t;
typedef __attribute__((ext_vector_type(16))) __bf16 v16bf;
typedef __attribute__((ext_vector_type(8)))  float  v8f;
typedef __attribute__((ext_vector_type(4)))  unsigned int v4u;
typedef __attribute__((ext_vector_type(8)))  int v8i_;
typedef __attribute__((ext_vector_type(4)))  int v4i_;

#if __has_builtin(__builtin_amdgcn_tensor_load_to_lds) && \
    __has_builtin(__builtin_amdgcn_s_wait_tensorcnt)
#define USE_TDM 1
#if __has_include(<hip/amd_detail/amd_gfx1250_TDM.h>)
#define TDM_6ARG 1
#endif
#endif

static constexpr int Bb = 8;
static constexpr int Ss = 2048;
static constexpr int Ee = 1024;

static constexpr int BM = 128, BN = 128, BK = 32;
static constexpr int LDT = BK + 8;  // LDS row stride: 32 elems (64B) + 4-DWORD pad

__device__ __forceinline__ unsigned short f2bf(float f) {
    unsigned int u = __float_as_uint(f);
    unsigned int r = u + 0x7FFFu + ((u >> 16) & 1u);   // round-to-nearest-even
    return (unsigned short)(r >> 16);
}

// ---------------------------------------------------------------- convert
__global__ void k_cvt_bf16(const float* __restrict__ in,
                           unsigned short* __restrict__ out, int n) {
    int i = blockIdx.x * blockDim.x + threadIdx.x;
    if (i < n) out[i] = f2bf(in[i]);
}

#ifdef USE_TDM
// Issue one TDM 2D tile load: 128 rows x 32 bf16 cols from row-major source
// (leading dim `ld` elements) into LDS at byte address `lds_byte`, with LDS
// padding of 4 DWORDs every 16 DWORDs (reproduces the LDT=40 layout).
__device__ __forceinline__ void tdm_tile(const unsigned short* gbase,
                                         long row0, int ld, int kt,
                                         unsigned lds_byte) {
    unsigned long long ga = (unsigned long long)(uintptr_t)gbase +
                            ((unsigned long long)((long)row0 * ld + kt) << 1);
    v4u g0;
    g0.x = 1u;                                   // count=1, user descriptor
    g0.y = lds_byte;                             // D#.lds_addr
    g0.z = (unsigned)ga;                         // D#.global_addr[31:0]
    g0.w = ((unsigned)(ga >> 32) & 0x01FFFFFFu)  // D#.global_addr[56:32]
           | 0x80000000u;                        // type=2 ("image")
    const unsigned td0 = (unsigned)ld;           // tensor_dim0 = row length
    const unsigned td1 = 1u << 30;               // tensor_dim1: large bound
    v8i_ g1;
    g1[0] = (int)((1u << 16)       // data_size = 1 -> 2 bytes
                  | (1u << 20)     // pad_enable
                  | (3u << 22)     // pad_interval = 16 DWORDs
                  | (3u << 25));   // pad_amount  = 4 DWORDs
    g1[1] = (int)((td0 & 0xFFFFu) << 16);                        // dim0 lo16
    g1[2] = (int)(((td0 >> 16) & 0xFFFFu) | ((td1 & 0xFFFFu) << 16));
    g1[3] = (int)(((td1 >> 16) & 0xFFFFu) | (32u << 16));        // tile_dim0=32
    g1[4] = (int)(128u);                                         // tile_dim1=128
    g1[5] = (int)td0;                                            // dim0 stride lo
    g1[6] = 0;
    g1[7] = 0;
    v4i_ gz = {0, 0, 0, 0};
#ifdef TDM_6ARG
    v8i_ gz8 = {0, 0, 0, 0, 0, 0, 0, 0};
    __builtin_amdgcn_tensor_load_to_lds(g0, g1, gz, gz, gz8, 0);
#else
    __builtin_amdgcn_tensor_load_to_lds(g0, g1, gz, gz, 0);
#endif
}
#endif  // USE_TDM

// ---------------------------------------------------------------- GEMM (NT)
// C[M,N] = A[M,K] * Bw[N,K]^T   (both bf16 row-major, K-major rows)
// 256 threads = 8 waves; block tile 128x128x32; wave tile 32x64 (2x4 WMMA).
template <bool BIAS, bool RELU, bool STF32, bool STBF, bool STBT>
__global__ __launch_bounds__(256) void k_gemm_nt(
    const unsigned short* __restrict__ A,
    const unsigned short* __restrict__ Bw,
    const float* __restrict__ bias,
    float* __restrict__ Cf,
    unsigned short* __restrict__ Cb,
    unsigned short* __restrict__ Ct,
    int M, int N, int K,
    int lda, int ldb, int ldc, int ldct,
    long sA, long sB, long sC, long sCt) {
    (void)M; (void)N;
    const int b = blockIdx.z;
    A  += (long)b * sA;
    Bw += (long)b * sB;

    const int tid   = threadIdx.x;
    const int lane  = tid & 31;
    const int wv    = tid >> 5;           // 0..7
    const int waveM = (wv & 3) * 32;      // 4 waves stacked in M
    const int waveN = (wv >> 2) * 64;     // 2 waves across N
    const int m0 = blockIdx.y * BM;
    const int n0 = blockIdx.x * BN;

    __shared__ alignas(16) unsigned short sAB[2][2][BM * LDT];

    v8f acc[2][4];
#pragma unroll
    for (int mt = 0; mt < 2; ++mt)
#pragma unroll
        for (int nt = 0; nt < 4; ++nt)
#pragma unroll
            for (int i = 0; i < 8; ++i) acc[mt][nt][i] = 0.0f;

    const int r16 = lane & 15;
    const int kh  = lane >> 4;   // K-half select per ISA 16-bit A/B layouts

#ifdef USE_TDM
    // ---------------- TDM double-buffered pipeline ----------------
    const unsigned ldsA0 = (unsigned)(size_t)&sAB[0][0][0];
    const unsigned ldsB0 = (unsigned)(size_t)&sAB[0][1][0];
    const unsigned ldsA1 = (unsigned)(size_t)&sAB[1][0][0];
    const unsigned ldsB1 = (unsigned)(size_t)&sAB[1][1][0];

    if (wv == 0) {
        tdm_tile(A,  m0, lda, 0, ldsA0);
        tdm_tile(Bw, n0, ldb, 0, ldsB0);
        __builtin_amdgcn_s_wait_tensorcnt(0);
    }
    __syncthreads();

    for (int kt = 0; kt < K; kt += BK) {
        const int buf = (kt / BK) & 1;
        const bool more = (kt + BK) < K;
        if (wv == 0 && more) {   // DMA next tiles into the back buffer
            tdm_tile(A,  m0, lda, kt + BK, buf ? ldsA0 : ldsA1);
            tdm_tile(Bw, n0, ldb, kt + BK, buf ? ldsB0 : ldsB1);
        }

        union Frag { uint4 q[2]; v16bf v; };
        Frag fa[2], fb[4];
#pragma unroll
        for (int mt = 0; mt < 2; ++mt) {
            const unsigned short* base =
                &sAB[buf][0][(waveM + mt * 16 + r16) * LDT];
            fa[mt].q[0] = *(const uint4*)&base[kh * 8];
            fa[mt].q[1] = *(const uint4*)&base[16 + kh * 8];
        }
#pragma unroll
        for (int nt = 0; nt < 4; ++nt) {
            const unsigned short* base =
                &sAB[buf][1][(waveN + nt * 16 + r16) * LDT];
            fb[nt].q[0] = *(const uint4*)&base[kh * 8];
            fb[nt].q[1] = *(const uint4*)&base[16 + kh * 8];
        }
#pragma unroll
        for (int mt = 0; mt < 2; ++mt)
#pragma unroll
            for (int nt = 0; nt < 4; ++nt)
                acc[mt][nt] = __builtin_amdgcn_wmma_f32_16x16x32_bf16(
                    false, fa[mt].v, false, fb[nt].v,
                    (short)0, acc[mt][nt], false, false);

        if (wv == 0 && more) __builtin_amdgcn_s_wait_tensorcnt(0);
        __syncthreads();
    }
#else
    // ---------------- VGPR-staging fallback ----------------
    auto loadTile = [&](int buf, int kt) {
#pragma unroll
        for (int p = 0; p < 2; ++p) {
            int idx = (p * 256 + tid) * 8;
            int r = idx >> 5;
            int c = idx & 31;
            const uint4* ga = (const uint4*)&A[(long)(m0 + r) * lda + kt + c];
            *(uint4*)&sAB[buf][0][r * LDT + c] = *ga;
            const uint4* gb = (const uint4*)&Bw[(long)(n0 + r) * ldb + kt + c];
            *(uint4*)&sAB[buf][1][r * LDT + c] = *gb;
            if (kt + 2 * BK < K) {
                __builtin_prefetch(&A [(long)(m0 + r) * lda + kt + 2 * BK + c], 0, 0);
                __builtin_prefetch(&Bw[(long)(n0 + r) * ldb + kt + 2 * BK + c], 0, 0);
            }
        }
    };

    loadTile(0, 0);
    for (int kt = 0; kt < K; kt += BK) {
        const int buf = (kt / BK) & 1;
        __syncthreads();
        if (kt + BK < K) loadTile(buf ^ 1, kt + BK);

        union Frag { uint4 q[2]; v16bf v; };
        Frag fa[2], fb[4];
#pragma unroll
        for (int mt = 0; mt < 2; ++mt) {
            const unsigned short* base =
                &sAB[buf][0][(waveM + mt * 16 + r16) * LDT];
            fa[mt].q[0] = *(const uint4*)&base[kh * 8];
            fa[mt].q[1] = *(const uint4*)&base[16 + kh * 8];
        }
#pragma unroll
        for (int nt = 0; nt < 4; ++nt) {
            const unsigned short* base =
                &sAB[buf][1][(waveN + nt * 16 + r16) * LDT];
            fb[nt].q[0] = *(const uint4*)&base[kh * 8];
            fb[nt].q[1] = *(const uint4*)&base[16 + kh * 8];
        }
#pragma unroll
        for (int mt = 0; mt < 2; ++mt)
#pragma unroll
            for (int nt = 0; nt < 4; ++nt)
                acc[mt][nt] = __builtin_amdgcn_wmma_f32_16x16x32_bf16(
                    false, fa[mt].v, false, fb[nt].v,
                    (short)0, acc[mt][nt], false, false);
    }
#endif

    // Epilogue: C/D layout -> VGPR i: M = i (+8 for lanes 16..31), N = lane%16
#pragma unroll
    for (int mt = 0; mt < 2; ++mt) {
#pragma unroll
        for (int nt = 0; nt < 4; ++nt) {
#pragma unroll
            for (int vi = 0; vi < 8; ++vi) {
                int row = m0 + waveM + mt * 16 + vi + ((lane >> 4) << 3);
                int col = n0 + waveN + nt * 16 + (lane & 15);
                float v = acc[mt][nt][vi];
                if (BIAS) v += bias[col];
                if (RELU) v = v > 0.0f ? v : 0.0f;
                if (STF32) Cf[(long)b * sC + (long)row * ldc + col] = v;
                if (STBF)  Cb[(long)b * sC + (long)row * ldc + col] = f2bf(v);
                if (STBT)  Ct[(long)b * sCt + (long)col * ldct + row] = f2bf(v);
            }
        }
    }
}

// ---------------------------------------------------------------- lengths
__global__ __launch_bounds__(256) void k_lengths(const float* __restrict__ energy,
                                                 int* __restrict__ len) {
    int b = blockIdx.x;
    const float* row = energy + (long)b * Ss * Ss;
    int cnt = 0;
    for (int j = threadIdx.x; j < Ss; j += 256) cnt += (row[j] != 0.0f) ? 1 : 0;
    for (int m = 16; m; m >>= 1) cnt += __shfl_xor(cnt, m, 32);
    __shared__ int sred[8];
    if ((threadIdx.x & 31) == 0) sred[threadIdx.x >> 5] = cnt;
    __syncthreads();
    if (threadIdx.x == 0) {
        int t = 0;
#pragma unroll
        for (int i = 0; i < 8; ++i) t += sred[i];
        len[b] = t;
    }
}

// ---------------------------------------------------------------- softmax
__global__ __launch_bounds__(256) void k_softmax(const float* __restrict__ energy,
                                                 const int* __restrict__ len,
                                                 unsigned short* __restrict__ score) {
    const long rowid = blockIdx.x;
    const int  b = (int)(rowid / Ss);
    const float* e = energy + rowid * Ss;
    unsigned short* o = score + rowid * Ss;
    const int L = len[b];

    float v[8];
    float mx = -3.402823466e38f;
#pragma unroll
    for (int i = 0; i < 8; ++i) {
        int j = threadIdx.x + i * 256;
        float x = e[j] * 0.125f;            // 1/sqrt(dk), dk = 64
        if (j >= L) x = -__builtin_inff();
        v[i] = x;
        mx = fmaxf(mx, x);
    }
    for (int m = 16; m; m >>= 1) mx = fmaxf(mx, __shfl_xor(mx, m, 32));
    __shared__ float smax[8], ssum[8];
    if ((threadIdx.x & 31) == 0) smax[threadIdx.x >> 5] = mx;
    __syncthreads();
    mx = smax[0];
#pragma unroll
    for (int i = 1; i < 8; ++i) mx = fmaxf(mx, smax[i]);

    float sum = 0.0f;
#pragma unroll
    for (int i = 0; i < 8; ++i) { float ex = __expf(v[i] - mx); v[i] = ex; sum += ex; }
    for (int m = 16; m; m >>= 1) sum += __shfl_xor(sum, m, 32);
    if ((threadIdx.x & 31) == 0) ssum[threadIdx.x >> 5] = sum;
    __syncthreads();
    sum = 0.0f;
#pragma unroll
    for (int i = 0; i < 8; ++i) sum += ssum[i];
    float inv = 1.0f / sum;
#pragma unroll
    for (int i = 0; i < 8; ++i) {
        int j = threadIdx.x + i * 256;
        o[j] = f2bf(v[i] * inv);
    }
}

// ---------------------------------------------------------------- add + LN
template <bool STBF>
__global__ __launch_bounds__(256) void k_add_ln(const float* __restrict__ x,
                                                const float* __restrict__ z,
                                                const float* __restrict__ g,
                                                const float* __restrict__ beta,
                                                float* __restrict__ out,
                                                unsigned short* __restrict__ outb) {
    const long row = blockIdx.x;
    const int  t = threadIdx.x;
    const float4 xv = ((const float4*)(x + row * Ee))[t];
    const float4 zv = ((const float4*)(z + row * Ee))[t];
    float s0 = xv.x + zv.x, s1 = xv.y + zv.y, s2 = xv.z + zv.z, s3 = xv.w + zv.w;
    float sum = s0 + s1 + s2 + s3;
    float sq  = s0 * s0 + s1 * s1 + s2 * s2 + s3 * s3;
    for (int m = 16; m; m >>= 1) {
        sum += __shfl_xor(sum, m, 32);
        sq  += __shfl_xor(sq,  m, 32);
    }
    __shared__ float r1[8], r2[8];
    if ((t & 31) == 0) { r1[t >> 5] = sum; r2[t >> 5] = sq; }
    __syncthreads();
    sum = 0.0f; sq = 0.0f;
#pragma unroll
    for (int i = 0; i < 8; ++i) { sum += r1[i]; sq += r2[i]; }
    const float mean = sum * (1.0f / Ee);
    const float var  = sq * (1.0f / Ee) - mean * mean;
    const float rstd = rsqrtf(var + 1e-5f);

    const float4 gv = ((const float4*)g)[t];
    const float4 bv = ((const float4*)beta)[t];
    float o0 = (s0 - mean) * rstd * gv.x + bv.x;
    float o1 = (s1 - mean) * rstd * gv.y + bv.y;
    float o2 = (s2 - mean) * rstd * gv.z + bv.z;
    float o3 = (s3 - mean) * rstd * gv.w + bv.w;
    ((float4*)(out + row * Ee))[t] = make_float4(o0, o1, o2, o3);
    if (STBF) {
        ushort4 h = make_ushort4(f2bf(o0), f2bf(o1), f2bf(o2), f2bf(o3));
        ((ushort4*)(outb + row * Ee))[t] = h;
    }
}

// ---------------------------------------------------------------- launch
extern "C" void kernel_launch(void* const* d_in, const int* in_sizes, int n_in,
                              void* d_out, int out_size, void* d_ws, size_t ws_size,
                              hipStream_t stream) {
    (void)in_sizes; (void)n_in; (void)out_size; (void)ws_size;
    const float* x     = (const float*)d_in[0];
    const float* Wq    = (const float*)d_in[1];
    const float* Wk    = (const float*)d_in[2];
    const float* Wv    = (const float*)d_in[3];
    const float* fc1_w = (const float*)d_in[4];
    const float* fc1_b = (const float*)d_in[5];
    const float* fc2_w = (const float*)d_in[6];
    const float* fc2_b = (const float*)d_in[7];
    const float* ln1_g = (const float*)d_in[8];
    const float* ln1_b = (const float*)d_in[9];
    const float* ln2_g = (const float*)d_in[10];
    const float* ln2_b = (const float*)d_in[11];

    const long BS = (long)Bb * Ss;   // 16384 rows

    char* ws = (char*)d_ws;
    size_t off = 0;
    auto carve = [&](size_t bytes) -> char* {
        char* p = ws + off;
        off = (off + bytes + 255) & ~(size_t)255;
        return p;
    };
    unsigned short* xb    = (unsigned short*)carve((size_t)BS * Ee * 2);
    unsigned short* wqb   = (unsigned short*)carve((size_t)Ee * Ee * 2);
    unsigned short* wkb   = (unsigned short*)carve((size_t)Ee * Ee * 2);
    unsigned short* wvb   = (unsigned short*)carve((size_t)Ee * Ee * 2);
    unsigned short* w1b   = (unsigned short*)carve((size_t)Ee * Ee * 2);
    unsigned short* w2b   = (unsigned short*)carve((size_t)Ee * Ee * 2);
    unsigned short* Qb    = (unsigned short*)carve((size_t)BS * Ee * 2);
    unsigned short* Kb    = (unsigned short*)carve((size_t)BS * Ee * 2);
    unsigned short* Vt    = (unsigned short*)carve((size_t)BS * Ee * 2); // [E][B*S]
    float*          energ = (float*)carve((size_t)BS * Ss * 4);
    unsigned short* score = (unsigned short*)carve((size_t)BS * Ss * 2);
    int*            lens  = (int*)carve(256);
    float*          zbuf  = (float*)carve((size_t)BS * Ee * 4);
    float*          x1    = (float*)carve((size_t)BS * Ee * 4);
    unsigned short* x1b   = (unsigned short*)carve((size_t)BS * Ee * 2);
    unsigned short* hbuf  = (unsigned short*)carve((size_t)BS * Ee * 2);
    float*          z2    = (float*)carve((size_t)BS * Ee * 4);

    {
        long nx = BS * Ee;
        k_cvt_bf16<<<(nx + 255) / 256, 256, 0, stream>>>(x, xb, (int)nx);
        int nw = Ee * Ee;
        int gw = (nw + 255) / 256;
        k_cvt_bf16<<<gw, 256, 0, stream>>>(Wq, wqb, nw);
        k_cvt_bf16<<<gw, 256, 0, stream>>>(Wk, wkb, nw);
        k_cvt_bf16<<<gw, 256, 0, stream>>>(Wv, wvb, nw);
        k_cvt_bf16<<<gw, 256, 0, stream>>>(fc1_w, w1b, nw);
        k_cvt_bf16<<<gw, 256, 0, stream>>>(fc2_w, w2b, nw);
    }

    dim3 gProj(Ee / BN, (unsigned)(BS / BM), 1);          // 8 x 128
    k_gemm_nt<false, false, false, true, false><<<gProj, 256, 0, stream>>>(
        xb, wqb, nullptr, nullptr, Qb, nullptr,
        (int)BS, Ee, Ee, Ee, Ee, Ee, 0, 0, 0, 0, 0);
    k_gemm_nt<false, false, false, true, false><<<gProj, 256, 0, stream>>>(
        xb, wkb, nullptr, nullptr, Kb, nullptr,
        (int)BS, Ee, Ee, Ee, Ee, Ee, 0, 0, 0, 0, 0);
    k_gemm_nt<false, false, false, false, true><<<gProj, 256, 0, stream>>>(
        xb, wvb, nullptr, nullptr, nullptr, Vt,
        (int)BS, Ee, Ee, Ee, Ee, 0, (int)BS, 0, 0, 0, 0);

    dim3 gE(Ss / BN, Ss / BM, Bb);                        // 16 x 16 x 8
    k_gemm_nt<false, false, true, false, false><<<gE, 256, 0, stream>>>(
        Qb, Kb, nullptr, energ, nullptr, nullptr,
        Ss, Ss, Ee, Ee, Ee, Ss, 0,
        (long)Ss * Ee, (long)Ss * Ee, (long)Ss * Ss, 0);

    k_lengths<<<Bb, 256, 0, stream>>>(energ, lens);
    k_softmax<<<(unsigned)BS, 256, 0, stream>>>(energ, lens, score);

    dim3 gZ(Ee / BN, Ss / BM, Bb);                        // 8 x 16 x 8
    k_gemm_nt<false, false, true, false, false><<<gZ, 256, 0, stream>>>(
        score, Vt, nullptr, zbuf, nullptr, nullptr,
        Ss, Ee, Ss, Ss, (int)BS, Ee, 0,
        (long)Ss * Ss, (long)Ss, (long)Ss * Ee, 0);

    k_add_ln<true><<<(unsigned)BS, 256, 0, stream>>>(x, zbuf, ln1_g, ln1_b, x1, x1b);

    k_gemm_nt<true, true, false, true, false><<<gProj, 256, 0, stream>>>(
        x1b, w1b, fc1_b, nullptr, hbuf, nullptr,
        (int)BS, Ee, Ee, Ee, Ee, Ee, 0, 0, 0, 0, 0);
    k_gemm_nt<true, false, true, false, false><<<gProj, 256, 0, stream>>>(
        hbuf, w2b, fc2_b, z2, nullptr, nullptr,
        (int)BS, Ee, Ee, Ee, Ee, Ee, 0, 0, 0, 0, 0);

    k_add_ln<false><<<(unsigned)BS, 256, 0, stream>>>(
        x1, z2, ln2_g, ln2_b, (float*)d_out, nullptr);
}
